// LSTM_30391188587313
// MI455X (gfx1250) — matrix operand order repaired and checked by reading, compiled, not run on hardware
//
#include <hip/hip_runtime.h>
#include <hip/hip_bf16.h>
#include <math.h>

// ---------------------------------------------------------------------------
// LSTM on MI455X (gfx1250): bf16 WMMA, persistent recurrent kernel with
// LDS-resident recurrent weights and register-resident cell state.
// B=128, S=512, I=512, H=1024, O=128
// ---------------------------------------------------------------------------

typedef __attribute__((ext_vector_type(16))) __bf16 v16bf;
typedef __attribute__((ext_vector_type(8)))  __bf16 v8bf;
typedef __attribute__((ext_vector_type(4)))  __bf16 v4bf;
typedef __attribute__((ext_vector_type(8)))  float  v8f;

#define B_  128
#define S_  512
#define I_  512
#define H_  1024
#define O_  128
#define NWG 64          // one WG per 16 hidden columns (x4 gates)
#define NT  256         // 8 waves of 32
#define KP  1536        // packed weight row: 512 (W_x) + 1024 (W_h)
#define LDS_ROW_BYTES 2064  // 1024 bf16 payload + 16B pad (stride % 64dw == 4)

// ---- WMMA helpers ---------------------------------------------------------

static __device__ __forceinline__ v8f wmma_bf16(v16bf a, v16bf b, v8f c) {
  // v_wmma_f32_16x16x32_bf16  (8 args: neg_a, A, neg_b, B, c_mod, C, reuseA, reuseB)
  return __builtin_amdgcn_wmma_f32_16x16x32_bf16(false, a, false, b, (short)0, c,
                                                 false, false);
}

// A fragment, 16x32 bf16, row-major source: lane holds 16 contiguous K values
// (caller passes p = row + k + half*16).
static __device__ __forceinline__ v16bf ld_a(const __bf16* p) {
  v8bf lo = *(const v8bf*)(p);
  v8bf hi = *(const v8bf*)(p + 8);
  return __builtin_shufflevector(lo, hi, 0,1,2,3,4,5,6,7,8,9,10,11,12,13,14,15);
}

// B fragment, 32x16 bf16: lane holds 16 contiguous K values of row n
// (caller passes p = W[n] + k + half*16).
static __device__ __forceinline__ v16bf ld_b(const __bf16* p) {
  v8bf lo = *(const v8bf*)(p);
  v8bf hi = *(const v8bf*)(p + 8);
  return __builtin_shufflevector(lo, hi, 0,1,2,3,4,5,6,7,8,9,10,11,12,13,14,15);
}

static __device__ __forceinline__ float sigmoidf_(float x) {
  return 1.0f / (1.0f + __expf(-x));
}

// ---- conversion / packing kernels ----------------------------------------

__global__ void cvt_x_bf16(const float4* __restrict__ in, v4bf* __restrict__ out,
                           int n4) {
  int i = blockIdx.x * blockDim.x + threadIdx.x;
  if (i < n4) {
    float4 v = in[i];
    v4bf o;
    o[0] = (__bf16)v.x; o[1] = (__bf16)v.y; o[2] = (__bf16)v.z; o[3] = (__bf16)v.w;
    out[i] = o;
  }
}

// wpack[g*H + n][0:512]  = W_x[g][n][:]   (bf16)
// wpack[g*H + n][512:1536] = W_h[g][n][:] (bf16)
// biaspack[g*H + n] = b_x[g][n] + b_h[g][n]
// whyp[row][0:1024] = W_hy[row][:] (bf16)
__global__ void pack_weights(const float* __restrict__ wxi, const float* __restrict__ whi,
                             const float* __restrict__ wxf, const float* __restrict__ whf,
                             const float* __restrict__ wxc, const float* __restrict__ whc,
                             const float* __restrict__ wxo, const float* __restrict__ who,
                             const float* __restrict__ bxi, const float* __restrict__ bhi,
                             const float* __restrict__ bxf, const float* __restrict__ bhf,
                             const float* __restrict__ bxc, const float* __restrict__ bhc,
                             const float* __restrict__ bxo, const float* __restrict__ bho,
                             const float* __restrict__ why,
                             __bf16* __restrict__ wpack, __bf16* __restrict__ whyp,
                             float* __restrict__ biaspack) {
  const int p = blockIdx.x;
  const int tid = threadIdx.x;
  if (p < 4 * H_) {
    const int g = p >> 10, nn = p & (H_ - 1);
    const float *wx, *wh, *bx, *bh;
    if      (g == 0) { wx = wxi; wh = whi; bx = bxi; bh = bhi; }
    else if (g == 1) { wx = wxf; wh = whf; bx = bxf; bh = bhf; }
    else if (g == 2) { wx = wxc; wh = whc; bx = bxc; bh = bhc; }
    else             { wx = wxo; wh = who; bx = bxo; bh = bho; }
    __bf16* dst = wpack + (size_t)p * KP;
    for (int k = tid; k < I_; k += NT) dst[k]       = (__bf16)wx[(size_t)nn * I_ + k];
    for (int k = tid; k < H_; k += NT) dst[I_ + k]  = (__bf16)wh[(size_t)nn * H_ + k];
    if (tid == 0) biaspack[p] = bx[nn] + bh[nn];
  } else {
    const int row = p - 4 * H_;
    for (int k = tid; k < H_; k += NT)
      whyp[(size_t)row * H_ + k] = (__bf16)why[(size_t)row * H_ + k];
  }
}

// ---- grid-wide barrier (64 persistent WGs) --------------------------------

static __device__ __forceinline__ void grid_barrier(unsigned* bar, unsigned target) {
  __syncthreads();
  if (threadIdx.x == 0) {
    __hip_atomic_fetch_add(bar, 1u, __ATOMIC_ACQ_REL, __HIP_MEMORY_SCOPE_AGENT);
    while (__hip_atomic_load(bar, __ATOMIC_ACQUIRE, __HIP_MEMORY_SCOPE_AGENT) < target) {
      __builtin_amdgcn_s_sleep(2);
    }
  }
  __syncthreads();
}

// ---- persistent recurrent kernel ------------------------------------------
// WG wg owns hidden columns [wg*16, wg*16+16) of all 4 gates.
// Wave w owns batch rows [16w, 16w+16). Cell state c lives in VGPRs.

__global__ void __launch_bounds__(NT, 1)
lstm_recurrent(const __bf16* __restrict__ xbf,     // (B, S, I) bf16
               const __bf16* __restrict__ wpack,   // (4H, 1536) bf16
               const float*  __restrict__ biaspack,// (4H) f32
               __bf16* __restrict__ h0,            // (B, H) bf16 ping
               __bf16* __restrict__ h1,            // (B, H) bf16 pong
               float* __restrict__ out_h,          // (B, H) f32
               float* __restrict__ out_c,          // (B, H) f32
               unsigned* __restrict__ bar) {
  extern __shared__ char smem[];                   // 64 x LDS_ROW_BYTES

  const int tid  = threadIdx.x;
  const int wave = tid >> 5;
  const int lane = tid & 31;
  const int half = lane >> 4;
  const int r    = lane & 15;
  const int wg   = blockIdx.x;
  const int nb   = wg * 16;
  const int n    = nb + r;                         // hidden column of this lane

  // Stage this WG's W_h block (4 gates x 16 cols x 1024 K, bf16) into LDS once.
  for (int idx = tid; idx < 64 * 128; idx += NT) {           // 16B chunks
    const int row = idx >> 7;                                // 0..63 = g*16+col
    const int off = (idx & 127) * 8;                         // bf16 elements
    const int g = row >> 4, col = row & 15;
    const v8bf* src =
        (const v8bf*)(wpack + (size_t)(g * H_ + nb + col) * KP + I_ + off);
    *(v8bf*)(&smem[row * LDS_ROW_BYTES + off * 2]) = *src;
  }
  __syncthreads();

  float bg[4];
  #pragma unroll
  for (int g = 0; g < 4; ++g) bg[g] = biaspack[g * H_ + n];

  const int m  = 16 * wave + r;          // A-fragment row this lane loads
  const int m2 = 16 * wave + half * 8;   // C-fragment row base for this lane
  const __bf16* xrow = xbf + (size_t)m * (S_ * I_);

  float cst[8], hlast[8];
  #pragma unroll
  for (int i = 0; i < 8; ++i) { cst[i] = 0.0f; hlast[i] = 0.0f; }

  for (int t = 0; t < S_; ++t) {
    const __bf16* hread  = (t & 1) ? h1 : h0;
    __bf16*       hwrite = (t & 1) ? h0 : h1;

    v8f acc[4];
    #pragma unroll
    for (int g = 0; g < 4; ++g) {
      #pragma unroll
      for (int e = 0; e < 8; ++e) acc[g][e] = bg[g];
    }

    // ---- x-projection part: K = 512, B operand from L2-resident wpack
    const __bf16* xa = xrow + (size_t)t * I_;
    if (t + 1 < S_) __builtin_prefetch(xrow + (size_t)(t + 1) * I_ + lane * 16, 0, 1);
    #pragma unroll 2
    for (int k = 0; k < I_; k += 32) {
      v16bf a = ld_a(xa + k + half * 16);
      #pragma unroll
      for (int g = 0; g < 4; ++g) {
        v16bf b = ld_b(wpack + (size_t)(g * H_ + n) * KP + k + half * 16);
        acc[g] = wmma_bf16(a, b, acc[g]);
      }
    }

    // ---- h-projection part: K = 1024, B operand from LDS
    const __bf16* ha = hread + (size_t)m * H_;
    #pragma unroll 2
    for (int kk = 0; kk < H_; kk += 32) {
      v16bf a = ld_a(ha + kk + half * 16);
      #pragma unroll
      for (int g = 0; g < 4; ++g) {
        const __bf16* sp =
            (const __bf16*)&smem[(g * 16 + r) * LDS_ROW_BYTES + (kk + half * 16) * 2];
        acc[g] = wmma_bf16(a, ld_b(sp), acc[g]);
      }
    }

    // ---- gate nonlinearities + state update: i/f/g/o are in the same lane
    #pragma unroll
    for (int rr = 0; rr < 8; ++rr) {
      float iv = sigmoidf_(acc[0][rr]);
      float fv = sigmoidf_(acc[1][rr]);
      float gv = tanhf(acc[2][rr]);
      float ov = sigmoidf_(acc[3][rr]);
      float c  = fv * cst[rr] + iv * gv;
      cst[rr]  = c;
      float h  = ov * tanhf(c);
      hlast[rr] = h;
      hwrite[(size_t)(m2 + rr) * H_ + n] = (__bf16)h;
    }

    __threadfence();
    grid_barrier(bar, (unsigned)(NWG * (t + 1)));
  }

  #pragma unroll
  for (int rr = 0; rr < 8; ++rr) {
    out_h[(size_t)(m2 + rr) * H_ + n] = hlast[rr];
    out_c[(size_t)(m2 + rr) * H_ + n] = cst[rr];
  }
}

// ---- final projection y = h @ W_hy^T + b_hy -------------------------------

__global__ void __launch_bounds__(NT)
output_proj(const __bf16* __restrict__ hbuf, const __bf16* __restrict__ whyp,
            const float* __restrict__ bhy, float* __restrict__ y) {
  const int tid  = threadIdx.x;
  const int wave = tid >> 5;
  const int lane = tid & 31;
  const int half = lane >> 4;
  const int r    = lane & 15;
  const int n    = blockIdx.x * 16 + r;
  const int m    = 16 * wave + r;
  const int m2   = 16 * wave + half * 8;

  v8f acc;
  #pragma unroll
  for (int e = 0; e < 8; ++e) acc[e] = 0.0f;

  const __bf16* ha = hbuf + (size_t)m * H_;
  const __bf16* wr = whyp + (size_t)n * H_;
  #pragma unroll 4
  for (int kk = 0; kk < H_; kk += 32) {
    v16bf a = ld_a(ha + kk + half * 16);
    v16bf b = ld_b(wr + kk + half * 16);
    acc = wmma_bf16(a, b, acc);
  }
  const float bb = bhy[n];
  #pragma unroll
  for (int rr = 0; rr < 8; ++rr)
    y[(size_t)(m2 + rr) * O_ + n] = acc[rr] + bb;
}

// ---------------------------------------------------------------------------

extern "C" void kernel_launch(void* const* d_in, const int* in_sizes, int n_in,
                              void* d_out, int out_size, void* d_ws, size_t ws_size,
                              hipStream_t stream) {
  (void)in_sizes; (void)n_in; (void)out_size; (void)ws_size;

  const float* x   = (const float*)d_in[0];
  const float* Wxi = (const float*)d_in[1];  const float* bxi = (const float*)d_in[2];
  const float* Whi = (const float*)d_in[3];  const float* bhi = (const float*)d_in[4];
  const float* Wxf = (const float*)d_in[5];  const float* bxf = (const float*)d_in[6];
  const float* Whf = (const float*)d_in[7];  const float* bhf = (const float*)d_in[8];
  const float* Wxc = (const float*)d_in[9];  const float* bxc = (const float*)d_in[10];
  const float* Whc = (const float*)d_in[11]; const float* bhc = (const float*)d_in[12];
  const float* Wxo = (const float*)d_in[13]; const float* bxo = (const float*)d_in[14];
  const float* Who = (const float*)d_in[15]; const float* bho = (const float*)d_in[16];
  const float* Why = (const float*)d_in[17]; const float* bhy = (const float*)d_in[18];

  // workspace layout (bytes)
  char* ws = (char*)d_ws;
  const size_t off_bar  = 0;                       // 256 B (counter + pad)
  const size_t off_h0   = 256;                     // 262144 B
  const size_t off_h1   = off_h0 + (size_t)B_ * H_ * 2;
  const size_t off_x    = off_h1 + (size_t)B_ * H_ * 2;          // 64 MB
  const size_t off_wp   = off_x  + (size_t)B_ * S_ * I_ * 2;     // 12 MB
  const size_t off_whyp = off_wp + (size_t)4 * H_ * KP * 2;      // 256 KB
  const size_t off_bias = off_whyp + (size_t)O_ * H_ * 2;        // 16 KB

  unsigned* bar   = (unsigned*)(ws + off_bar);
  __bf16*   h0    = (__bf16*)(ws + off_h0);
  __bf16*   h1    = (__bf16*)(ws + off_h1);
  __bf16*   xbf   = (__bf16*)(ws + off_x);
  __bf16*   wpack = (__bf16*)(ws + off_wp);
  __bf16*   whyp  = (__bf16*)(ws + off_whyp);
  float*    biasp = (float*)(ws + off_bias);

  float* y_out = (float*)d_out;                    // (B, O)
  float* h_out = y_out + (size_t)B_ * O_;          // (B, H)
  float* c_out = h_out + (size_t)B_ * H_;          // (B, H)

  // zero barrier counter + both h buffers (h at t=0 is zero)
  hipMemsetAsync(d_ws, 0, off_x, stream);

  const int n4 = (B_ * S_ * I_) / 4;
  cvt_x_bf16<<<n4 / 256, 256, 0, stream>>>((const float4*)x, (v4bf*)xbf, n4);

  pack_weights<<<4 * H_ + O_, NT, 0, stream>>>(
      Wxi, Whi, Wxf, Whf, Wxc, Whc, Wxo, Who,
      bxi, bhi, bxf, bhf, bxc, bhc, bxo, bho,
      Why, wpack, whyp, biasp);

  lstm_recurrent<<<NWG, NT, 64 * LDS_ROW_BYTES, stream>>>(
      xbf, wpack, biasp, h0, h1, h_out, c_out, bar);

  // after 512 steps the final h lives in h0 (t=511 writes buf[(511+1)&1]=h0)
  output_proj<<<O_ / 16, NT, 0, stream>>>(h0, whyp, bhy, y_out);
}
